// pointnet_17918603559577
// MI455X (gfx1250) — compile-verified, matrix operand order checked by reading, not run on hardware
//
#include <hip/hip_runtime.h>
#include <hip/hip_bf16.h>

#define BATCH 8
#define NPTS  4096

typedef __attribute__((ext_vector_type(16))) _Float16 v16h;
typedef __attribute__((ext_vector_type(8)))  _Float16 v8h;
typedef __attribute__((ext_vector_type(8)))  float    v8f;

static inline int pad32i(int x){ return (x + 31) & ~31; }

// ---------------------------------------------------------------------------
// xyz [B,3,N] -> xyz_nl [B,N,3] f32 and l0h [B,N,3] f16
// ---------------------------------------------------------------------------
__global__ void prep_xyz(const float* __restrict__ xyz, float* __restrict__ xn,
                         _Float16* __restrict__ xh, int total) {
    int t = blockIdx.x * blockDim.x + threadIdx.x;
    if (t >= total) return;                       // total = B*N
    int b = t / NPTS, n = t % NPTS;
    #pragma unroll
    for (int c = 0; c < 3; c++) {
        float v = xyz[((size_t)b * 3 + c) * NPTS + n];
        xn[(size_t)t * 3 + c] = v;
        xh[(size_t)t * 3 + c] = (_Float16)v;
    }
}

// ---------------------------------------------------------------------------
// Farthest point sampling. One block per batch. Matches reference: idx[0]=0,
// argmax ties -> lowest index.
// ---------------------------------------------------------------------------
#define FPS_T 256
__global__ void fps_kernel(const float* __restrict__ xyz, // [B,N,3]
                           float* __restrict__ dist,      // [B,N] scratch
                           int* __restrict__ idxOut,      // [B,S]
                           int N, int S) {
    int b = blockIdx.x, t = threadIdx.x;
    const float* xb = xyz + (size_t)b * N * 3;
    float* db = dist + (size_t)b * N;
    for (int i = t; i < N; i += FPS_T) db[i] = 1e10f;
    __shared__ float sd[FPS_T];
    __shared__ int   si[FPS_T];
    __shared__ float cent[3];
    int far = 0;
    if (t == 0) idxOut[(size_t)b * S] = 0;
    __syncthreads();
    for (int s = 1; s < S; s++) {
        if (t == 0) { cent[0]=xb[(size_t)far*3]; cent[1]=xb[(size_t)far*3+1]; cent[2]=xb[(size_t)far*3+2]; }
        __syncthreads();
        float cx = cent[0], cy = cent[1], cz = cent[2];
        float best = -1.0f; int bi = 0;
        for (int i = t; i < N; i += FPS_T) {
            float dx = xb[(size_t)i*3]   - cx;
            float dy = xb[(size_t)i*3+1] - cy;
            float dz = xb[(size_t)i*3+2] - cz;
            float d  = dx*dx + dy*dy + dz*dz;
            float od = db[i];
            float nd = d < od ? d : od;
            db[i] = nd;
            if (nd > best) { best = nd; bi = i; }   // strided i increasing -> first max kept
        }
        sd[t] = best; si[t] = bi;
        __syncthreads();
        for (int off = FPS_T / 2; off > 0; off >>= 1) {
            if (t < off) {
                if (sd[t+off] > sd[t] || (sd[t+off] == sd[t] && si[t+off] < si[t])) {
                    sd[t] = sd[t+off]; si[t] = si[t+off];
                }
            }
            __syncthreads();
        }
        far = si[0];
        if (t == 0) idxOut[(size_t)b * S + s] = far;
        __syncthreads();
    }
}

__global__ void gather_xyz(const float* __restrict__ xyz, const int* __restrict__ idx,
                           float* __restrict__ out, int N, int S, int total) {
    int t = blockIdx.x * blockDim.x + threadIdx.x;
    if (t >= total) return;                       // total = B*S
    int b = t / S;
    int id = idx[t];
    out[(size_t)t*3+0] = xyz[((size_t)b*N + id)*3+0];
    out[(size_t)t*3+1] = xyz[((size_t)b*N + id)*3+1];
    out[(size_t)t*3+2] = xyz[((size_t)b*N + id)*3+2];
}

// ---------------------------------------------------------------------------
// Ball query: first K indices (ascending) with d2<=r2, fill with first hit.
// ---------------------------------------------------------------------------
__global__ void ball_query(const float* __restrict__ xyz,  // [B,N,3]
                           const float* __restrict__ nxyz, // [B,S,3]
                           int* __restrict__ gi,           // [B,S,K]
                           int N, int S, int K, float r2, int total) {
    int t = blockIdx.x * blockDim.x + threadIdx.x;
    if (t >= total) return;                       // total = B*S
    int b = t / S;
    const float* c = nxyz + (size_t)t * 3;
    const float* xb = xyz + (size_t)b * N * 3;
    int* g = gi + (size_t)t * K;
    float cx = c[0], cy = c[1], cz = c[2];
    int cnt = 0, first = 0;
    for (int i = 0; i < N && cnt < K; i++) {
        float dx = xb[(size_t)i*3]   - cx;
        float dy = xb[(size_t)i*3+1] - cy;
        float dz = xb[(size_t)i*3+2] - cz;
        float d2 = dx*dx + dy*dy + dz*dz;
        if (d2 <= r2) { if (cnt == 0) first = i; g[cnt++] = i; }
    }
    for (int k = cnt; k < K; k++) g[k] = first;
}

// ---------------------------------------------------------------------------
// Group: out[(b*S+s)*K+k, :] = [feats[b,idx,:D], xyz[b,idx]-new_xyz[b,s]], pad 0
// ---------------------------------------------------------------------------
__global__ void group_concat(const float* __restrict__ xyz,      // [B,N,3]
                             const float* __restrict__ nxyz,     // [B,S,3]
                             const _Float16* __restrict__ feats, // [B,N,D]
                             const int* __restrict__ gi,         // [B,S,K]
                             _Float16* __restrict__ out,         // [R,Cp]
                             int N, int S, int K, int D, int Cp, int total) {
    int t = blockIdx.x * blockDim.x + threadIdx.x;
    if (t >= total) return;                       // total = B*S*K
    int ss = t / K;
    int b  = ss / S;
    int id = gi[t];
    const _Float16* f = feats + ((size_t)b * N + id) * D;
    _Float16* o = out + (size_t)t * Cp;
    for (int c = 0; c < D; c++) o[c] = f[c];
    const float* pc = nxyz + (size_t)ss * 3;
    const float* pp = xyz + ((size_t)b * N + id) * 3;
    o[D]   = (_Float16)(pp[0] - pc[0]);
    o[D+1] = (_Float16)(pp[1] - pc[1]);
    o[D+2] = (_Float16)(pp[2] - pc[2]);
    for (int c = D + 3; c < Cp; c++) o[c] = (_Float16)0.f;
}

// ---------------------------------------------------------------------------
// Fold BN into weights/bias: w16[o,c] = w[o,c]*gamma*rsqrt(var+eps) (fp16),
// bias[o] = (b-mean)*s + beta. Zero padding.
// ---------------------------------------------------------------------------
__global__ void fuse_bn(const float* __restrict__ w, const float* __restrict__ b,
                        const float* __restrict__ g, const float* __restrict__ be,
                        const float* __restrict__ m, const float* __restrict__ v,
                        _Float16* __restrict__ w16, float* __restrict__ bias,
                        int cout, int cin, int coutP, int cinP) {
    int t = blockIdx.x * blockDim.x + threadIdx.x;
    int tot = coutP * cinP;
    if (t >= tot) return;
    int o = t / cinP, c = t % cinP;
    if (o < cout) {
        float s = g[o] * rsqrtf(v[o] + 1e-5f);
        w16[t] = (c < cin) ? (_Float16)(w[(size_t)o * cin + c] * s) : (_Float16)0.f;
        if (c == 0) bias[o] = (b[o] - m[o]) * s + be[o];
    } else {
        w16[t] = (_Float16)0.f;
        if (c == 0) bias[o] = 0.f;
    }
}

// ---------------------------------------------------------------------------
// WMMA GEMM, 32x32 tile per wave, register double-buffered K pipeline.
// Y[r,o] = relu(sum_k X[r,k]*W[o,k] + bias[o]).
// A fragment per ISA layout: lane l holds row M=l&15; halves 0..7 = K kb..kb+7,
// halves 8..15 = K kb+16..kb+23, kb = 8*(l>>4). B symmetric over W rows (cols).
// C/D layout: VGPR v, lane l -> M = v + 8*(l>>4), N = l&15.
// ---------------------------------------------------------------------------
#define WMMA_MM(A, Bf, C) \
    __builtin_amdgcn_wmma_f32_16x16x32_f16(false, (A).v, false, (Bf).v, (short)0, (C), false, false)

__global__ void __launch_bounds__(32)
gemm_bias_relu(const _Float16* __restrict__ X,  // [R32, Kp]
               const _Float16* __restrict__ W,  // [Cp, Kp]
               const float* __restrict__ bias,  // [Cp]
               _Float16* __restrict__ Y,        // [R32, Cp]
               int Kp, int Cp) {
    int lane = threadIdx.x;
    int row0 = blockIdx.x * 32;
    int col0 = blockIdx.y * 32;
    int m  = lane & 15;
    int kb = (lane >> 4) * 8;
    const _Float16* x0 = X + (size_t)(row0 + m) * Kp + kb;
    const _Float16* x1 = x0 + (size_t)16 * Kp;
    const _Float16* w0 = W + (size_t)(col0 + m) * Kp + kb;
    const _Float16* w1 = w0 + (size_t)16 * Kp;

    union H16 { v16h v; v8h h[2]; };
    v8f c00 = {}, c01 = {}, c10 = {}, c11 = {};
    H16 a0, a1, b0, b1, na0, na1, nb0, nb1;

    // prologue: chunk k = 0
    a0.h[0] = *(const v8h*)(x0);      a0.h[1] = *(const v8h*)(x0 + 16);
    a1.h[0] = *(const v8h*)(x1);      a1.h[1] = *(const v8h*)(x1 + 16);
    b0.h[0] = *(const v8h*)(w0);      b0.h[1] = *(const v8h*)(w0 + 16);
    b1.h[0] = *(const v8h*)(w1);      b1.h[1] = *(const v8h*)(w1 + 16);

    for (int k = 32; k < Kp; k += 32) {
        // issue next chunk's loads before consuming current chunk
        na0.h[0] = *(const v8h*)(x0 + k);      na0.h[1] = *(const v8h*)(x0 + k + 16);
        na1.h[0] = *(const v8h*)(x1 + k);      na1.h[1] = *(const v8h*)(x1 + k + 16);
        nb0.h[0] = *(const v8h*)(w0 + k);      nb0.h[1] = *(const v8h*)(w0 + k + 16);
        nb1.h[0] = *(const v8h*)(w1 + k);      nb1.h[1] = *(const v8h*)(w1 + k + 16);
        if (k + 32 < Kp) {
            __builtin_prefetch(x0 + k + 32, 0, 1);
            __builtin_prefetch(x1 + k + 32, 0, 1);
            __builtin_prefetch(w0 + k + 32, 0, 1);
            __builtin_prefetch(w1 + k + 32, 0, 1);
        }
        c00 = WMMA_MM(a0, b0, c00);
        c01 = WMMA_MM(a0, b1, c01);
        c10 = WMMA_MM(a1, b0, c10);
        c11 = WMMA_MM(a1, b1, c11);
        a0 = na0; a1 = na1; b0 = nb0; b1 = nb1;
    }
    c00 = WMMA_MM(a0, b0, c00);
    c01 = WMMA_MM(a0, b1, c01);
    c10 = WMMA_MM(a1, b0, c10);
    c11 = WMMA_MM(a1, b1, c11);

    int n = lane & 15;
    int rbase = (lane >> 4) * 8;
    float bv0 = bias[col0 + n];
    float bv1 = bias[col0 + 16 + n];
    #pragma unroll
    for (int v = 0; v < 8; v++) {
        size_t rA = (size_t)(row0 + rbase + v) * Cp;
        size_t rB = (size_t)(row0 + 16 + rbase + v) * Cp;
        float f00 = c00[v] + bv0; f00 = f00 > 0.f ? f00 : 0.f;
        float f01 = c01[v] + bv1; f01 = f01 > 0.f ? f01 : 0.f;
        float f10 = c10[v] + bv0; f10 = f10 > 0.f ? f10 : 0.f;
        float f11 = c11[v] + bv1; f11 = f11 > 0.f ? f11 : 0.f;
        Y[rA + col0 + n]      = (_Float16)f00;
        Y[rA + col0 + 16 + n] = (_Float16)f01;
        Y[rB + col0 + n]      = (_Float16)f10;
        Y[rB + col0 + 16 + n] = (_Float16)f11;
    }
}

// ---------------------------------------------------------------------------
// Max over K samples: dst[b,s,off+c] = max_k Y[(b*S+s)*K+k, c]
// ---------------------------------------------------------------------------
__global__ void maxpool_k(const _Float16* __restrict__ Y, _Float16* __restrict__ dst,
                          int S, int K, int C, int Cp, int Dtot, int off, int total) {
    int t = blockIdx.x * blockDim.x + threadIdx.x;
    if (t >= total) return;                       // total = B*S*C
    int c = t % C, ss = t / C;
    const _Float16* y = Y + (size_t)ss * K * Cp + c;
    float mx = (float)y[0];
    for (int k = 1; k < K; k++) {
        float v = (float)y[(size_t)k * Cp];
        if (v > mx) mx = v;
    }
    dst[(size_t)ss * Dtot + off + c] = (_Float16)mx;
}

// ---------------------------------------------------------------------------
// SA3 group_all concat: rows b*128+s -> [l2_xyz(3), l2_points(512)] pad to 544
// ---------------------------------------------------------------------------
__global__ void sa3_concat(const float* __restrict__ nxyz2, const _Float16* __restrict__ l2p,
                           _Float16* __restrict__ out, int total) {
    int t = blockIdx.x * blockDim.x + threadIdx.x;
    if (t >= total) return;                       // total = 1024*544
    int r = t / 544, c = t % 544;
    _Float16 v = (_Float16)0.f;
    if (c < 3)        v = (_Float16)nxyz2[(size_t)r * 3 + c];
    else if (c < 515) v = l2p[(size_t)r * 512 + (c - 3)];
    out[t] = v;
}

// FP3 concat: rows b*128+s -> [l2_points(512), broadcast l3(1024)] = 1536
__global__ void fp3_concat(const _Float16* __restrict__ l2p, const _Float16* __restrict__ l3p,
                           _Float16* __restrict__ out, int total) {
    int t = blockIdx.x * blockDim.x + threadIdx.x;
    if (t >= total) return;                       // total = 1024*1536
    int r = t / 1536, c = t % 1536;
    int b = r / 128;
    out[t] = (c < 512) ? l2p[(size_t)r * 512 + c]
                       : l3p[(size_t)b * 1024 + (c - 512)];
}

// ---------------------------------------------------------------------------
// 3-NN inverse-distance interp + concat: out row b*N1+n = [p1(D1), interp(D2)],
// pad to Cp. Strict '<' keeps lowest index on ties (matches top_k stability).
// ---------------------------------------------------------------------------
__global__ void three_nn_concat(const float* __restrict__ xyz1,   // [B,N1,3]
                                const float* __restrict__ xyz2,   // [B,S2,3]
                                const _Float16* __restrict__ p1,  // [B,N1,D1]
                                const _Float16* __restrict__ p2,  // [B,S2,D2]
                                _Float16* __restrict__ out,       // [R,Cp]
                                int N1, int S2, int D1, int D2, int Cp, int total) {
    int t = blockIdx.x * blockDim.x + threadIdx.x;
    if (t >= total) return;                       // total = B*N1
    int b = t / N1;
    const float* q = xyz1 + (size_t)t * 3;
    const float* xb = xyz2 + (size_t)b * S2 * 3;
    float qx = q[0], qy = q[1], qz = q[2];
    float d0 = 1e30f, d1 = 1e30f, d2 = 1e30f;
    int   i0 = 0, i1 = 0, i2 = 0;
    for (int j = 0; j < S2; j++) {
        float dx = xb[(size_t)j*3]   - qx;
        float dy = xb[(size_t)j*3+1] - qy;
        float dz = xb[(size_t)j*3+2] - qz;
        float dd = dx*dx + dy*dy + dz*dz;
        if (dd < d0)      { d2=d1; i2=i1; d1=d0; i1=i0; d0=dd; i0=j; }
        else if (dd < d1) { d2=d1; i2=i1; d1=dd; i1=j; }
        else if (dd < d2) { d2=dd; i2=j; }
    }
    float w0 = 1.f / (d0 + 1e-8f);
    float w1 = 1.f / (d1 + 1e-8f);
    float w2 = 1.f / (d2 + 1e-8f);
    float ws = w0 + w1 + w2;
    w0 /= ws; w1 /= ws; w2 /= ws;
    const _Float16* f0 = p2 + ((size_t)b * S2 + i0) * D2;
    const _Float16* f1 = p2 + ((size_t)b * S2 + i1) * D2;
    const _Float16* f2 = p2 + ((size_t)b * S2 + i2) * D2;
    const _Float16* s1 = p1 + (size_t)t * D1;
    _Float16* o = out + (size_t)t * Cp;
    for (int c = 0; c < D1; c++) o[c] = s1[c];
    for (int c = 0; c < D2; c++) {
        float v = w0 * (float)f0[c] + w1 * (float)f1[c] + w2 * (float)f2[c];
        o[D1 + c] = (_Float16)v;
    }
    for (int c = D1 + D2; c < Cp; c++) o[c] = (_Float16)0.f;
}

__global__ void copy_h16(const _Float16* __restrict__ in, _Float16* __restrict__ out, int total) {
    int t = blockIdx.x * blockDim.x + threadIdx.x;
    if (t < total) out[t] = in[t];
}

// final: Y [B,N,256] f16 -> d_out [B,256,N] f32
__global__ void final_out(const _Float16* __restrict__ Y, float* __restrict__ out, int total) {
    int t = blockIdx.x * blockDim.x + threadIdx.x;
    if (t >= total) return;                       // total = B*256*N
    int n = t % NPTS;
    int r = t / NPTS;
    int c = r % 256, b = r / 256;
    out[t] = (float)Y[((size_t)b * NPTS + n) * 256 + c];
}

// ---------------------------------------------------------------------------
// Host orchestration
// ---------------------------------------------------------------------------
struct Layer { const float *w, *b, *g, *be, *m, *v; int cin, cout; };

extern "C" void kernel_launch(void* const* d_in, const int* in_sizes, int n_in,
                              void* d_out, int out_size, void* d_ws, size_t ws_size,
                              hipStream_t stream) {
    (void)in_sizes; (void)n_in; (void)out_size; (void)ws_size;
    const float* xyz = (const float*)d_in[0];

    // --- parameter walk (dict insertion order; per-layer: w,b,gamma,beta,mean,var)
    Layer L[24]; int li = 0, ii = 1;
    auto grab = [&](int cin, const int* ch, int n) {
        for (int j = 0; j < n; j++) {
            Layer& l = L[li++];
            l.w  = (const float*)d_in[ii++];
            l.b  = (const float*)d_in[ii++];
            l.g  = (const float*)d_in[ii++];
            l.be = (const float*)d_in[ii++];
            l.m  = (const float*)d_in[ii++];
            l.v  = (const float*)d_in[ii++];
            l.cin = cin; l.cout = ch[j]; cin = ch[j];
        }
    };
    const int sa1_0[] = {32, 32, 64},  sa1_1[] = {64, 64, 128}, sa1_2[] = {64, 96, 128};
    const int sa2_0[] = {128, 128, 256}, sa2_1[] = {128, 196, 256};
    const int sa3c[]  = {256, 512, 1024};
    const int fpc[]   = {256, 256};
    grab(6, sa1_0, 3); grab(6, sa1_1, 3); grab(6, sa1_2, 3);
    grab(323, sa2_0, 3); grab(323, sa2_1, 3);
    grab(515, sa3c, 3);
    grab(1536, fpc, 2); grab(576, fpc, 2); grab(259, fpc, 2);

    // --- workspace bump allocator
    char* wp = (char*)d_ws;
    auto alloc = [&](size_t bytes) -> void* {
        void* r = (void*)wp; wp += (bytes + 255) & ~(size_t)255; return r;
    };
    float*     xyz_nl = (float*)alloc((size_t)BATCH * NPTS * 3 * 4);
    _Float16*  l0h    = (_Float16*)alloc((size_t)BATCH * NPTS * 3 * 2);
    float*     dist   = (float*)alloc((size_t)BATCH * NPTS * 4);
    int*       idx1   = (int*)alloc((size_t)BATCH * 512 * 4);
    int*       idx2   = (int*)alloc((size_t)BATCH * 128 * 4);
    float*     nxyz1  = (float*)alloc((size_t)BATCH * 512 * 3 * 4);
    float*     nxyz2  = (float*)alloc((size_t)BATCH * 128 * 3 * 4);
    int*       gi     = (int*)alloc((size_t)BATCH * 512 * 128 * 4);
    _Float16*  l1p    = (_Float16*)alloc((size_t)BATCH * 512 * 320 * 2);
    _Float16*  l2p    = (_Float16*)alloc((size_t)BATCH * 128 * 512 * 2);
    _Float16*  l2pn   = (_Float16*)alloc((size_t)BATCH * 128 * 256 * 2);
    _Float16*  l1pn   = (_Float16*)alloc((size_t)BATCH * 512 * 256 * 2);
    _Float16*  l3p    = (_Float16*)alloc((size_t)BATCH * 1024 * 2);
    _Float16*  wbuf   = (_Float16*)alloc((size_t)1024 * 1536 * 2);
    float*     bias   = (float*)alloc(1024 * 4);
    const size_t BUFE = (size_t)BATCH * 512 * 128 * 128;  // max activation elems
    _Float16*  bufA   = (_Float16*)alloc(BUFE * 2);
    _Float16*  bufB   = (_Float16*)alloc(BUFE * 2);

    const int TB = 256;
    auto blks = [&](long long n) { return (unsigned)((n + TB - 1) / TB); };

    // run an MLP chain of nl layers starting at L[ls]; input act in bufA/bufB.
    // All R in this network are multiples of 32; channel pads are multiples of 32.
    auto run_mlp = [&](int ls, int nl, int R, _Float16* actIn, int cp) -> _Float16* {
        int R32 = (R + 31) & ~31;
        _Float16* cur = actIn;
        _Float16* nxt = (actIn == bufA) ? bufB : bufA;
        for (int j = 0; j < nl; j++) {
            Layer& l = L[ls + j];
            int cinP = pad32i(l.cin);
            int coutP = pad32i(l.cout);
            int tot = coutP * cinP;
            fuse_bn<<<blks(tot), TB, 0, stream>>>(l.w, l.b, l.g, l.be, l.m, l.v,
                                                  wbuf, bias, l.cout, l.cin, coutP, cinP);
            dim3 gg((unsigned)(R32 / 32), (unsigned)(coutP / 32));
            gemm_bias_relu<<<gg, dim3(32), 0, stream>>>(cur, wbuf, bias, nxt, cp, coutP);
            cur = nxt; cp = coutP;
            nxt = (cur == bufA) ? bufB : bufA;
        }
        return cur;
    };

    // --- level 0 prep
    prep_xyz<<<blks((long long)BATCH * NPTS), TB, 0, stream>>>(xyz, xyz_nl, l0h, BATCH * NPTS);

    // --- SA1: FPS 512, three scales
    fps_kernel<<<BATCH, FPS_T, 0, stream>>>(xyz_nl, dist, idx1, NPTS, 512);
    gather_xyz<<<blks((long long)BATCH * 512), TB, 0, stream>>>(xyz_nl, idx1, nxyz1, NPTS, 512, BATCH * 512);
    {
        const float rr[3] = {0.1f, 0.2f, 0.4f};
        const int   kk[3] = {32, 64, 128};
        const int   ls[3] = {0, 3, 6};
        const int   co[3] = {64, 128, 128};
        const int   of[3] = {0, 64, 192};
        for (int sc = 0; sc < 3; sc++) {
            int K = kk[sc], R = BATCH * 512 * K;
            ball_query<<<blks((long long)BATCH * 512), TB, 0, stream>>>(
                xyz_nl, nxyz1, gi, NPTS, 512, K, rr[sc] * rr[sc], BATCH * 512);
            group_concat<<<blks((long long)R), TB, 0, stream>>>(
                xyz_nl, nxyz1, l0h, gi, bufA, NPTS, 512, K, 3, 32, R);
            _Float16* act = run_mlp(ls[sc], 3, R, bufA, 32);
            maxpool_k<<<blks((long long)BATCH * 512 * co[sc]), TB, 0, stream>>>(
                act, l1p, 512, K, co[sc], pad32i(co[sc]), 320, of[sc], BATCH * 512 * co[sc]);
        }
    }

    // --- SA2: FPS 128 on l1, two scales (feats = l1p, D=320 -> Cp 352)
    fps_kernel<<<BATCH, FPS_T, 0, stream>>>(nxyz1, dist, idx2, 512, 128);
    gather_xyz<<<blks((long long)BATCH * 128), TB, 0, stream>>>(nxyz1, idx2, nxyz2, 512, 128, BATCH * 128);
    {
        const float rr[2] = {0.4f, 0.8f};
        const int   kk[2] = {64, 128};
        const int   ls[2] = {9, 12};
        const int   of[2] = {0, 256};
        for (int sc = 0; sc < 2; sc++) {
            int K = kk[sc], R = BATCH * 128 * K;
            ball_query<<<blks((long long)BATCH * 128), TB, 0, stream>>>(
                nxyz1, nxyz2, gi, 512, 128, K, rr[sc] * rr[sc], BATCH * 128);
            group_concat<<<blks((long long)R), TB, 0, stream>>>(
                nxyz1, nxyz2, l1p, gi, bufA, 512, 128, K, 320, 352, R);
            _Float16* act = run_mlp(ls[sc], 3, R, bufA, 352);
            maxpool_k<<<blks((long long)BATCH * 128 * 256), TB, 0, stream>>>(
                act, l2p, 128, K, 256, 256, 512, of[sc], BATCH * 128 * 256);
        }
    }

    // --- SA3: group_all -> [B*128, 544] -> MLP -> maxpool over 128 -> l3p [B,1024]
    {
        int R = BATCH * 128;
        sa3_concat<<<blks((long long)R * 544), TB, 0, stream>>>(nxyz2, l2p, bufA, R * 544);
        _Float16* act = run_mlp(15, 3, R, bufA, 544);
        maxpool_k<<<blks((long long)BATCH * 1024), TB, 0, stream>>>(
            act, l3p, 1, 128, 1024, 1024, 1024, 0, BATCH * 1024);
    }

    // --- FP3: concat [l2p(512), broadcast l3(1024)] -> MLP -> l2pn [B,128,256]
    {
        int R = BATCH * 128;
        fp3_concat<<<blks((long long)R * 1536), TB, 0, stream>>>(l2p, l3p, bufA, R * 1536);
        _Float16* act = run_mlp(18, 2, R, bufA, 1536);
        copy_h16<<<blks((long long)R * 256), TB, 0, stream>>>(act, l2pn, R * 256);
    }

    // --- FP2: 3-NN (l1_xyz <- l2_xyz) + concat [l1p(320), interp(256)] = 576
    {
        int R = BATCH * 512;
        three_nn_concat<<<blks((long long)R), TB, 0, stream>>>(
            nxyz1, nxyz2, l1p, l2pn, bufA, 512, 128, 320, 256, 576, R);
        _Float16* act = run_mlp(20, 2, R, bufA, 576);
        copy_h16<<<blks((long long)R * 256), TB, 0, stream>>>(act, l1pn, R * 256);
    }

    // --- FP1: 3-NN (l0 <- l1) + concat [l0(3), interp(256)] = 259 -> pad 288
    {
        int R = BATCH * NPTS;
        three_nn_concat<<<blks((long long)R), TB, 0, stream>>>(
            xyz_nl, nxyz1, l0h, l1pn, bufA, NPTS, 512, 3, 256, 288, R);
        _Float16* act = run_mlp(22, 2, R, bufA, 288);
        final_out<<<blks((long long)BATCH * 256 * NPTS), TB, 0, stream>>>(
            act, (float*)d_out, BATCH * 256 * NPTS);
    }
}
